// QuantiZ_73581379715441
// MI455X (gfx1250) — matrix operand order repaired and checked by reading, not compiled, not run on hardware
//
#include <hip/hip_runtime.h>
#include <hip/hip_bf16.h>

typedef __attribute__((ext_vector_type(16))) __bf16 v16bf;
typedef __attribute__((ext_vector_type(8)))  __bf16 v8bf;
typedef __attribute__((ext_vector_type(8)))  float  v8f;

#define NUM_CODE 4096
#define IN_DIM   1024
#define CODE_DIM 256
#define NROWS    32768       // 32*32*32
#define LDS_STRIDE 264       // 256 elems + 8 pad (528 B row: multiple of 16 B,
                             // rotates banks by 16 B per row -> conflict-light)

__device__ __forceinline__ __bf16 bf16_hi(float x) { return (__bf16)x; }
__device__ __forceinline__ __bf16 bf16_lo(float x) {
    return (__bf16)(x - (float)((__bf16)x));
}

__device__ __forceinline__ v8f wmma_bf16(v16bf a, v16bf b, v8f c) {
    return __builtin_amdgcn_wmma_f32_16x16x32_bf16(
        false, a, false, b, (short)0, c, false, false);
}

// Load 16 bf16 (32 B) from a 16B-aligned LDS address as two b128 reads.
__device__ __forceinline__ v16bf ld_b32x16(const __bf16* p) {
    v8bf a = *(const v8bf*)p;
    v8bf b = *(const v8bf*)(p + 8);
    v16bf r;
    #pragma unroll
    for (int i = 0; i < 8; ++i) { r[i] = a[i]; r[8 + i] = b[i]; }
    return r;
}

// ---------------------------------------------------------------------------
// Kernel 1: e = codebook @ proj_w.T + proj_b  (bf16x3 split WMMA, fp32 acc)
// One wave computes one 16x16 tile of e (4096 waves). Also emits bf16 hi/lo
// split of e for the distance GEMM's B operand.
// ---------------------------------------------------------------------------
__global__ void project_e_kernel(const float* __restrict__ codebook, // (4096,1024)
                                 const float* __restrict__ proj_w,   // (256,1024)
                                 const float* __restrict__ proj_b,   // (256)
                                 float*  __restrict__ e,             // (4096,256)
                                 __bf16* __restrict__ e_hi,          // (4096,256)
                                 __bf16* __restrict__ e_lo)          // (4096,256)
{
    const int lane  = threadIdx.x & 31;
    const int wave  = threadIdx.x >> 5;
    const int wgid  = blockIdx.x * 8 + wave;       // 512 blocks * 8 waves = 4096
    const int mTile = wgid >> 4;                   // 256 M-tiles (codes)
    const int nTile = wgid & 15;                   // 16  N-tiles (dims)
    const int laneL = lane & 15;
    const int laneH = lane >> 4;

    const int m = mTile * 16 + laneL;              // A row for this lane
    const int n = nTile * 16 + laneL;              // B column for this lane

    v8f acc = {};
    for (int kk = 0; kk < IN_DIM; kk += 32) {
        // A (16x32): lane holds K = kk+laneH*16+{0..15}
        const float* ap = codebook + (size_t)m * IN_DIM + kk + laneH * 16;
        v16bf a_hi, a_lo;
        #pragma unroll
        for (int j = 0; j < 8; ++j) {
            float x = ap[j];
            float y = ap[8 + j];
            a_hi[j]     = bf16_hi(x);  a_lo[j]     = bf16_lo(x);
            a_hi[8 + j] = bf16_hi(y);  a_lo[8 + j] = bf16_lo(y);
        }
        // B (32x16): B[k][n] = proj_w[n][k]; lane holds K = kk+laneH*16+{0..15}
        const float* bp = proj_w + (size_t)n * IN_DIM + kk + laneH * 16;
        v16bf b_hi, b_lo;
        #pragma unroll
        for (int j = 0; j < 16; ++j) {
            float x = bp[j];
            b_hi[j] = bf16_hi(x);  b_lo[j] = bf16_lo(x);
        }
        acc = wmma_bf16(a_hi, b_hi, acc);   // hi*hi
        acc = wmma_bf16(a_hi, b_lo, acc);   // hi*lo
        acc = wmma_bf16(a_lo, b_hi, acc);   // lo*hi
    }

    // C/D layout: lane holds column n, rows mTile*16 + laneH*8 + j
    const float bias = proj_b[n];
    #pragma unroll
    for (int j = 0; j < 8; ++j) {
        float v = acc[j] + bias;
        int row = mTile * 16 + laneH * 8 + j;
        size_t idx = (size_t)row * CODE_DIM + nTile * 16 + laneL;
        e[idx]    = v;
        e_hi[idx] = bf16_hi(v);
        e_lo[idx] = bf16_lo(v);
    }
}

// ---------------------------------------------------------------------------
// Kernel 2: per-code squared norm ||e[c]||^2. One wave per code.
// ---------------------------------------------------------------------------
__global__ void enorm_kernel(const float* __restrict__ e, float* __restrict__ enorm)
{
    const int gwave = (blockIdx.x * blockDim.x + threadIdx.x) >> 5; // code id
    const int lane  = threadIdx.x & 31;
    const float* row = e + (size_t)gwave * CODE_DIM;
    float s = 0.f;
    #pragma unroll
    for (int j = lane; j < CODE_DIM; j += 32) { float v = row[j]; s += v * v; }
    #pragma unroll
    for (int off = 16; off > 0; off >>= 1) s += __shfl_xor(s, off, 32);
    if (lane == 0) enorm[gwave] = s;
}

// ---------------------------------------------------------------------------
// Kernel 3: fused distance GEMM + argmin.
// metric(m, c) = ||e_c||^2 - 2 * <z_m, e_c>   (same argmin as the reference).
// One wave owns 16 z rows (A split once into registers). The 8 waves of a
// block sweep the same 256 code tiles, so B (16 codes x 256 dims, hi+lo) is
// staged in LDS ONCE per block (8x L2 traffic cut), double-buffered with one
// barrier per tile. 24 WMMAs per tile feed a running per-lane min, then a
// 16-lane shuffle reduction with first-index tiebreak.
// ---------------------------------------------------------------------------
__global__ void __launch_bounds__(256)
argmin_kernel(const float*  __restrict__ z,      // (32768,256)
              const __bf16* __restrict__ e_hi,   // (4096,256)
              const __bf16* __restrict__ e_lo,   // (4096,256)
              const float*  __restrict__ enorm,  // (4096)
              int* __restrict__ zidx)            // (32768)
{
    __shared__ __align__(16) __bf16 sB[2][2][16 * LDS_STRIDE]; // [buf][hi/lo][row*stride+col]

    const int tid   = threadIdx.x;
    const int lane  = tid & 31;
    const int wave  = tid >> 5;
    const int mTile = blockIdx.x * 8 + wave;   // 256 blocks * 8 waves = 2048
    const int laneL = lane & 15;
    const int laneH = lane >> 4;
    const int m = mTile * 16 + laneL;

    // ---- Preload + split the 16x256 z tile: 8 K-chunks of 32 ----
    v16bf a_hi[8], a_lo[8];
    const float* ap = z + (size_t)m * CODE_DIM;
    #pragma unroll
    for (int c = 0; c < 8; ++c) {
        const int kk = c * 32 + laneH * 16;
        #pragma unroll
        for (int j = 0; j < 8; ++j) {
            float x = ap[kk + j];
            float y = ap[kk + 8 + j];
            a_hi[c][j]     = bf16_hi(x);  a_lo[c][j]     = bf16_lo(x);
            a_hi[c][8 + j] = bf16_hi(y);  a_lo[c][8 + j] = bf16_lo(y);
        }
    }

    // ---- Cooperative staging geometry: tile = 16 rows x 512 B per array.
    // Each of 256 threads moves two 16 B chunks per array (rows r0 and r0+8).
    const int r0   = tid >> 5;            // 0..7
    const int colE = (tid & 31) * 8;      // element column, 16 B aligned
    const size_t gOff0 = (size_t)r0 * CODE_DIM + colE;
    const int    sOff0 = r0 * LDS_STRIDE + colE;

    uint4 h0, h1, l0, l1;
    {   // prologue: tile 0
        const __bf16* gh = e_hi + gOff0;
        const __bf16* gl = e_lo + gOff0;
        h0 = *(const uint4*)gh;  h1 = *(const uint4*)(gh + 8 * CODE_DIM);
        l0 = *(const uint4*)gl;  l1 = *(const uint4*)(gl + 8 * CODE_DIM);
    }

    float bestVal[8];
    int   bestIdx[8];
    #pragma unroll
    for (int j = 0; j < 8; ++j) { bestVal[j] = 3.4e38f; bestIdx[j] = 0; }

    for (int nt = 0; nt < NUM_CODE / 16; ++nt) {
        const int buf = nt & 1;

        // publish tile nt into LDS
        *(uint4*)&sB[buf][0][sOff0]                  = h0;
        *(uint4*)&sB[buf][0][sOff0 + 8 * LDS_STRIDE] = h1;
        *(uint4*)&sB[buf][1][sOff0]                  = l0;
        *(uint4*)&sB[buf][1][sOff0 + 8 * LDS_STRIDE] = l1;
        __syncthreads();

        // start fetching tile nt+1 (overlaps with WMMA below)
        if (nt + 1 < NUM_CODE / 16) {
            const size_t g = (size_t)(nt + 1) * 16 * CODE_DIM + gOff0;
            __builtin_prefetch(e_hi + g + 16 * CODE_DIM, 0, 1); // tile nt+2
            __builtin_prefetch(e_lo + g + 16 * CODE_DIM, 0, 1);
            h0 = *(const uint4*)(e_hi + g);
            h1 = *(const uint4*)(e_hi + g + 8 * CODE_DIM);
            l0 = *(const uint4*)(e_lo + g);
            l1 = *(const uint4*)(e_lo + g + 8 * CODE_DIM);
        }

        // compute from LDS: lane reads column (code) laneL, K = laneH*16 + c*32
        const __bf16* bh = &sB[buf][0][laneL * LDS_STRIDE + laneH * 16];
        const __bf16* bl = &sB[buf][1][laneL * LDS_STRIDE + laneH * 16];
        v8f acc = {};
        #pragma unroll
        for (int c = 0; c < 8; ++c) {
            v16bf b_hi = ld_b32x16(bh + c * 32);
            v16bf b_lo = ld_b32x16(bl + c * 32);
            acc = wmma_bf16(a_hi[c], b_hi, acc);
            acc = wmma_bf16(a_hi[c], b_lo, acc);
            acc = wmma_bf16(a_lo[c], b_hi, acc);
        }

        const int   code = nt * 16 + laneL;
        const float n2   = enorm[code];
        #pragma unroll
        for (int j = 0; j < 8; ++j) {
            float metric = n2 - 2.0f * acc[j];   // row = mTile*16 + laneH*8 + j
            if (metric < bestVal[j]) { bestVal[j] = metric; bestIdx[j] = code; }
        }
        // single barrier per iteration is sufficient with 2 buffers (see header)
    }

    // Min-reduce across the 16 lanes of each half (they share the same 8 rows).
    #pragma unroll
    for (int off = 1; off < 16; off <<= 1) {
        #pragma unroll
        for (int j = 0; j < 8; ++j) {
            float ov = __shfl_xor(bestVal[j], off, 32);
            int   oi = __shfl_xor(bestIdx[j], off, 32);
            if (ov < bestVal[j] || (ov == bestVal[j] && oi < bestIdx[j])) {
                bestVal[j] = ov; bestIdx[j] = oi;
            }
        }
    }
    if (laneL == 0) {
        #pragma unroll
        for (int j = 0; j < 8; ++j)
            zidx[mTile * 16 + laneH * 8 + j] = bestIdx[j];
    }
}

// ---------------------------------------------------------------------------
// Kernel 4: quant[m, :] = e[zidx[m], :]   (e is L2-resident: 4 MB)
// ---------------------------------------------------------------------------
__global__ void gather_kernel(const float* __restrict__ e,
                              const int*   __restrict__ zidx,
                              float* __restrict__ quant)
{
    size_t i = (size_t)blockIdx.x * blockDim.x + threadIdx.x;
    int row = (int)(i >> 8);
    int d   = (int)(i & 255);
    quant[i] = e[(size_t)zidx[row] * CODE_DIM + d];
}

// ---------------------------------------------------------------------------
extern "C" void kernel_launch(void* const* d_in, const int* in_sizes, int n_in,
                              void* d_out, int out_size, void* d_ws, size_t ws_size,
                              hipStream_t stream)
{
    const float* encode   = (const float*)d_in[0]; // (32,32,32,256) = (32768,256)
    const float* codebook = (const float*)d_in[1]; // (4096,1024)
    const float* proj_w   = (const float*)d_in[2]; // (256,1024)
    const float* proj_b   = (const float*)d_in[3]; // (256)

    // Workspace layout (256B-aligned offsets):
    //   e      fp32  4096*256 = 4 MB
    //   e_hi   bf16  4096*256 = 2 MB
    //   e_lo   bf16  4096*256 = 2 MB
    //   enorm  fp32  4096     = 16 KB
    char* ws = (char*)d_ws;
    float*  e     = (float*)ws;
    __bf16* e_hi  = (__bf16*)(ws + (size_t)NUM_CODE * CODE_DIM * 4);
    __bf16* e_lo  = e_hi + (size_t)NUM_CODE * CODE_DIM;
    float*  enorm = (float*)(ws + (size_t)NUM_CODE * CODE_DIM * 4
                                + (size_t)NUM_CODE * CODE_DIM * 2 * 2);

    // Outputs: zidx (32768 int32 slots) then quant (32768*256 fp32), flat.
    int*   zidx  = (int*)d_out;
    float* quant = (float*)d_out + NROWS;

    project_e_kernel<<<512, 256, 0, stream>>>(codebook, proj_w, proj_b, e, e_hi, e_lo);
    enorm_kernel   <<<512, 256, 0, stream>>>(e, enorm);
    argmin_kernel  <<<256, 256, 0, stream>>>(encode, e_hi, e_lo, enorm, zidx);
    gather_kernel  <<<NROWS, 256, 0, stream>>>(e, zidx, quant);
}
